// DynamicChunkEncoder_54829552500954
// MI455X (gfx1250) — matrix operand order repaired and checked by reading, not compiled, not run on hardware
//
#include <hip/hip_runtime.h>
#include <hip/hip_bf16.h>

typedef __attribute__((ext_vector_type(16))) __bf16 v16bf;
typedef __attribute__((ext_vector_type(8)))  float  v8f;
typedef int gv4i __attribute__((vector_size(16)));   // matches builtin param

union Frag { v16bf v; uint4 u[2]; };

constexpr int Bz   = 4;
constexpr int Lq   = 2048;
constexpr int Dm   = 512;
constexpr int Hh   = 8;
constexpr int HDim = 64;
constexpr int FFNd = 2048;
constexpr int SCH  = 512;     // attention chunk size for L=2048

constexpr int BM = 128, BN = 128, BK = 32;

#if __has_builtin(__builtin_amdgcn_global_load_async_to_lds_b128) && \
    __has_builtin(__builtin_amdgcn_s_wait_asynccnt)
#define USE_ASYNC 1
#else
#define USE_ASYNC 0
#endif

// 16B global->LDS copy: async DMA path on gfx1250, fallback = VGPR round trip
__device__ __forceinline__ void cp_b128(const void* g, void* l) {
#if USE_ASYNC
  __builtin_amdgcn_global_load_async_to_lds_b128(
      (__attribute__((address_space(1))) gv4i*)(void*)g,
      (__attribute__((address_space(3))) gv4i*)l, 0, 0);
#else
  *(uint4*)l = *(const uint4*)g;
#endif
}
__device__ __forceinline__ void cp_commit() {
#if USE_ASYNC
  __builtin_amdgcn_s_wait_asynccnt(0);
#endif
}

__device__ __forceinline__ v8f wmma_bf16(const Frag& a, const Frag& b, v8f c) {
  return __builtin_amdgcn_wmma_f32_16x16x32_bf16(false, a.v, false, b.v,
                                                 (short)0, c, false, false);
}

__device__ __forceinline__ float gelu_exact(float x) {
  return 0.5f * x * (1.0f + erff(x * 0.70710678118654752f));
}

__device__ __forceinline__ __bf16 to_bf16(float f) { return (__bf16)f; }
__device__ __forceinline__ __bf16 to_bf16(__bf16 f) { return f; }

// -------- reductions over 16-lane halves (match WMMA C row layout) --------
__device__ __forceinline__ float half16_max(float t) {
  t = fmaxf(t, __shfl_xor(t, 1));
  t = fmaxf(t, __shfl_xor(t, 2));
  t = fmaxf(t, __shfl_xor(t, 4));
  t = fmaxf(t, __shfl_xor(t, 8));
  return t;
}
__device__ __forceinline__ float half16_sum(float t) {
  t += __shfl_xor(t, 1);
  t += __shfl_xor(t, 2);
  t += __shfl_xor(t, 4);
  t += __shfl_xor(t, 8);
  return t;
}
__device__ __forceinline__ float wave_sum(float t) {
  t += __shfl_xor(t, 1);
  t += __shfl_xor(t, 2);
  t += __shfl_xor(t, 4);
  t += __shfl_xor(t, 8);
  t += __shfl_xor(t, 16);
  return t;
}

// -------- A-tile staging: manual (f32 -> bf16) or async DMA (bf16) --------
__device__ __forceinline__ void stage_a(const float* __restrict__ A,
                                        __bf16* sAbuf, int tid, int m0, int K,
                                        int k0) {
  #pragma unroll
  for (int it = 0; it < (BM * BK) / 256; ++it) {
    int idx = tid + it * 256;
    int r = idx >> 5, c = idx & 31;
    sAbuf[r * BK + c] = to_bf16(A[(size_t)(m0 + r) * K + (k0 + c)]);
  }
}
__device__ __forceinline__ void stage_a(const __bf16* __restrict__ A,
                                        __bf16* sAbuf, int tid, int m0, int K,
                                        int k0) {
  // rows are 32 contiguous bf16 = 4 x b128 per row; 512 chunks total
  #pragma unroll
  for (int it = 0; it < 2; ++it) {
    int chunk = tid + it * 256;
    int r = chunk >> 2;
    int c8 = (chunk & 3) * 8;
    cp_b128(A + (size_t)(m0 + r) * K + (k0 + c8), sAbuf + r * BK + c8);
  }
}

// ===========================================================================
// Tiled WMMA GEMM:  C[M,N] = A[M,K] @ B[K,N] + bias  (+ epilogue by MODE)
// MODE 0: QKV  -> split into q/k/v bf16 buffers laid out [B*H][L][HD]
// MODE 1: +resid, f32 out
// MODE 2: GELU, bf16 out
// Block 256 thr = 8 waves (4x2), block tile 128x128x32, wave tile 32x64.
// Double-buffered LDS, one barrier per K step.
// ===========================================================================
template <int MODE, typename AT>
__global__ __launch_bounds__(256) void gemm_kernel(
    const AT* __restrict__ A, const float* __restrict__ Bw,
    const float* __restrict__ bias, const float* __restrict__ resid,
    float* __restrict__ outF, __bf16* __restrict__ outH0,
    __bf16* __restrict__ outH1, __bf16* __restrict__ outH2,
    int M, int N, int K) {
  __shared__ alignas(16) __bf16 sA[2][BM * BK];   // [row][k]
  __shared__ alignas(16) __bf16 sB[2][BN * BK];   // [col][k] (K-contig/col)

  const int tid  = threadIdx.x;
  const int lane = tid & 31;
  const int wave = tid >> 5;
  const int m0 = blockIdx.y * BM;
  const int n0 = blockIdx.x * BN;
  const int wm = (wave >> 1) * 32;   // wave row offset in block tile
  const int wn = (wave & 1) * 64;    // wave col offset in block tile

  const int half8  = (lane < 16) ? 0 : 8;
  const int half16 = (lane < 16) ? 0 : 16;
  const int l15 = lane & 15;

  auto stage_b = [&](__bf16* sBbuf, int k0) {
    #pragma unroll
    for (int it = 0; it < (BK * BN) / 256; ++it) {
      int idx = tid + it * 256;
      int r = idx >> 7, c = idx & 127;
      sBbuf[c * BK + r] = (__bf16)Bw[(size_t)(k0 + r) * N + (n0 + c)];
    }
  };

  const int nk = K / BK;
  v8f acc[2][4] = {};

  stage_a(A, sA[0], tid, m0, K, 0);
  stage_b(sB[0], 0);

  for (int kt = 0; kt < nk; ++kt) {
    cp_commit();
    __syncthreads();
    if (kt + 1 < nk) {
      stage_a(A, sA[(kt + 1) & 1], tid, m0, K, (kt + 1) * BK);
      stage_b(sB[(kt + 1) & 1], (kt + 1) * BK);
    }
    const __bf16* cA = sA[kt & 1];
    const __bf16* cB = sB[kt & 1];

    Frag fa[2], fb[4];
    #pragma unroll
    for (int mi = 0; mi < 2; ++mi) {
      const __bf16* p = &cA[(wm + mi * 16 + l15) * BK + half8];
      fa[mi].u[0] = *(const uint4*)p;
      fa[mi].u[1] = *(const uint4*)(p + 16);
    }
    #pragma unroll
    for (int ni = 0; ni < 4; ++ni) {
      const __bf16* p = &cB[(wn + ni * 16 + l15) * BK + half8];
      fb[ni].u[0] = *(const uint4*)p;
      fb[ni].u[1] = *(const uint4*)(p + 16);
    }
    #pragma unroll
    for (int mi = 0; mi < 2; ++mi)
      #pragma unroll
      for (int ni = 0; ni < 4; ++ni)
        acc[mi][ni] = wmma_bf16(fa[mi], fb[ni], acc[mi][ni]);
  }

  // epilogue
  #pragma unroll
  for (int mi = 0; mi < 2; ++mi) {
    #pragma unroll
    for (int ni = 0; ni < 4; ++ni) {
      #pragma unroll
      for (int j = 0; j < 8; ++j) {
        int gr = m0 + wm + mi * 16 + ((lane < 16) ? j : j + 8);
        int gc = n0 + wn + ni * 16 + l15;
        float v = acc[mi][ni][j] + bias[gc];
        if (MODE == 1) {
          v += resid[(size_t)gr * N + gc];
          outF[(size_t)gr * N + gc] = v;
        } else if (MODE == 2) {
          outH0[(size_t)gr * N + gc] = (__bf16)gelu_exact(v);
        } else {  // QKV split: gc in [0,1536)
          int which = gc >> 9;
          int d = gc & 511;
          int h = d >> 6;
          int hd = d & 63;
          int bb = gr >> 11;       // row / L
          int l  = gr & 2047;      // row % L
          __bf16* dst = (which == 0) ? outH0 : (which == 1) ? outH1 : outH2;
          dst[(((size_t)(bb * Hh + h)) * Lq + l) * HDim + hd] = (__bf16)v;
        }
      }
    }
  }
}

// ===========================================================================
// Chunked local flash attention. Grid: (S/128, C, B*H). Block 256 = 8 waves.
// Each wave owns 16 query rows; key window streamed in 64-key tiles.
// Q/K tiles staged with async global->LDS DMA (bf16, layout-preserving).
// ===========================================================================
__global__ __launch_bounds__(256) void attn_kernel(
    const __bf16* __restrict__ q, const __bf16* __restrict__ k,
    const __bf16* __restrict__ v, const int* __restrict__ mask,
    __bf16* __restrict__ aout) {
  __shared__ alignas(16) __bf16 sQ[128 * HDim];     // [query][hd]
  __shared__ alignas(16) __bf16 sK[64 * HDim];      // [key][hd]
  __shared__ alignas(16) __bf16 sVt[HDim * 64];     // [hd][key]
  __shared__ alignas(16) __bf16 sP[8][16 * 64];     // per-wave P staging
  __shared__ float sMsk[64];

  const int bh   = blockIdx.z;
  const int ch   = blockIdx.y;
  const int qb   = blockIdx.x;
  const int tid  = threadIdx.x;
  const int lane = tid & 31;
  const int wave = tid >> 5;
  const int l15  = lane & 15;
  const int half8  = (lane < 16) ? 0 : 8;
  const int half16 = (lane < 16) ? 0 : 16;

  const int q0 = ch * SCH + qb * 128;
  const __bf16* qbase = q + (size_t)bh * Lq * HDim;
  const __bf16* kbase = k + (size_t)bh * Lq * HDim;
  const __bf16* vbase = v + (size_t)bh * Lq * HDim;
  const int bb = bh >> 3, hh = bh & 7;
  const int* mrow = mask + (size_t)bb * Lq;

  // stage Q tile: 128*64 bf16 = 1024 x b128
  {
    const __bf16* qg = qbase + (size_t)q0 * HDim;
    #pragma unroll
    for (int it = 0; it < 4; ++it) {
      int c = (tid + it * 256) * 8;
      cp_b128(qg + c, sQ + c);
    }
  }
  cp_commit();
  __syncthreads();

  // per-wave Q A-fragments (K = hd, 2 slices of 32)
  Frag aq[2];
  {
    int qrow = wave * 16 + l15;
    #pragma unroll
    for (int ks = 0; ks < 2; ++ks) {
      const __bf16* p = &sQ[qrow * HDim + ks * 32 + half8];
      aq[ks].u[0] = *(const uint4*)p;
      aq[ks].u[1] = *(const uint4*)(p + 16);
    }
  }

  float mrun[8], lrun[8];
  #pragma unroll
  for (int j = 0; j < 8; ++j) { mrun[j] = -INFINITY; lrun[j] = 0.f; }
  v8f O[4] = {};

  int kw0 = (ch - 1) * SCH; if (kw0 < 0) kw0 = 0;
  int kw1 = (ch + 2) * SCH; if (kw1 > Lq) kw1 = Lq;

  const float scale = 0.125f;  // 1/sqrt(64)

  for (int kt = kw0; kt < kw1; kt += 64) {
    // stage K tile (row-major == transposed-B layout for Q@K^T): 512 x b128
    {
      const __bf16* kg = kbase + (size_t)kt * HDim;
      #pragma unroll
      for (int it = 0; it < 2; ++it) {
        int c = (tid + it * 256) * 8;
        cp_b128(kg + c, sK + c);
      }
    }
    // stage V transposed: sVt[hd][key]
    #pragma unroll
    for (int it = 0; it < 16; ++it) {
      int idx = tid + it * 256;
      int kk = idx >> 6, hd = idx & 63;
      sVt[hd * 64 + kk] = vbase[(size_t)(kt + kk) * HDim + hd];
    }
    if (tid < 64) sMsk[tid] = mrow[kt + tid] ? -INFINITY : 0.0f;
    cp_commit();
    __syncthreads();

    // scores: 16 queries x 64 keys per wave (4 WMMA n-tiles x 2 k-steps)
    v8f sc[4];
    #pragma unroll
    for (int n = 0; n < 4; ++n) {
      v8f cc = {};
      #pragma unroll
      for (int ks = 0; ks < 2; ++ks) {
        Frag bk;
        const __bf16* p = &sK[(n * 16 + l15) * HDim + ks * 32 + half8];
        bk.u[0] = *(const uint4*)p;
        bk.u[1] = *(const uint4*)(p + 16);
        cc = wmma_bf16(aq[ks], bk, cc);
      }
      sc[n] = cc;
    }

    float mn[4];
    #pragma unroll
    for (int n = 0; n < 4; ++n) mn[n] = sMsk[n * 16 + l15];

    float p_[4][8];
    #pragma unroll
    for (int j = 0; j < 8; ++j) {
      float t = -INFINITY;
      #pragma unroll
      for (int n = 0; n < 4; ++n) {
        float s = sc[n][j] * scale + mn[n];
        p_[n][j] = s;
        t = fmaxf(t, s);
      }
      t = half16_max(t);
      float mnew = fmaxf(mrun[j], t);
      bool live = (mnew > -INFINITY);
      float alpha = live ? __expf(mrun[j] - mnew) : 0.f;
      float rs = 0.f;
      #pragma unroll
      for (int n = 0; n < 4; ++n) {
        float e = live ? __expf(p_[n][j] - mnew) : 0.f;
        p_[n][j] = e;
        rs += e;
      }
      rs = half16_sum(rs);
      lrun[j] = lrun[j] * alpha + rs;
      mrun[j] = mnew;
      #pragma unroll
      for (int nt = 0; nt < 4; ++nt) O[nt][j] *= alpha;
    }

    // write P (C-layout) into wave-private LDS as A-layout source [qrow][key]
    {
      int rj = (lane < 16) ? 0 : 8;
      #pragma unroll
      for (int n = 0; n < 4; ++n)
        #pragma unroll
        for (int j = 0; j < 8; ++j)
          sP[wave][(rj + j) * 64 + n * 16 + l15] = (__bf16)p_[n][j];
    }

    // O += P @ V   (A = P 16x64, B = V 64x64, 2 key-slices of 32)
    #pragma unroll
    for (int ks = 0; ks < 2; ++ks) {
      Frag ap;
      const __bf16* pp = &sP[wave][l15 * 64 + ks * 32 + half8];
      ap.u[0] = *(const uint4*)pp;
      ap.u[1] = *(const uint4*)(pp + 16);
      #pragma unroll
      for (int nt = 0; nt < 4; ++nt) {
        Frag bv;
        const __bf16* pv = &sVt[(nt * 16 + l15) * 64 + ks * 32 + half8];
        bv.u[0] = *(const uint4*)pv;
        bv.u[1] = *(const uint4*)(pv + 16);
        O[nt] = wmma_bf16(ap, bv, O[nt]);
      }
    }
    __syncthreads();
  }

  // normalize + write to a[B][L][D] (bf16), col = h*64 + hd
  #pragma unroll
  for (int nt = 0; nt < 4; ++nt) {
    #pragma unroll
    for (int j = 0; j < 8; ++j) {
      int row_l = q0 + wave * 16 + ((lane < 16) ? j : j + 8);
      float denom = lrun[j];
      float ov = (denom > 0.f) ? (O[nt][j] / denom) : 0.f;
      aout[((size_t)bb * Lq + row_l) * Dm + hh * HDim + nt * 16 + l15] =
          (__bf16)ov;
    }
  }
}

// ===========================================================================
// LayerNorm over D=512: one wave per row, 8 rows per block.
// ===========================================================================
__global__ __launch_bounds__(256) void ln_kernel(
    const float* __restrict__ x, const float* __restrict__ g,
    const float* __restrict__ b, float* __restrict__ out) {
  const int lane = threadIdx.x & 31;
  const int wave = threadIdx.x >> 5;
  const int row = blockIdx.x * 8 + wave;
  const float4* xv = (const float4*)(x + (size_t)row * Dm);

  float4 vals[4];
  float s = 0.f, ss = 0.f;
  #pragma unroll
  for (int i = 0; i < 4; ++i) {
    float4 t = xv[lane + i * 32];
    vals[i] = t;
    s += t.x + t.y + t.z + t.w;
    ss += t.x * t.x + t.y * t.y + t.z * t.z + t.w * t.w;
  }
  s = wave_sum(s);
  ss = wave_sum(ss);
  const float mu = s * (1.f / Dm);
  const float var = ss * (1.f / Dm) - mu * mu;
  const float rstd = rsqrtf(var + 1e-5f);

  float* orow = out + (size_t)row * Dm;
  #pragma unroll
  for (int i = 0; i < 4; ++i) {
    int c0 = (lane + i * 32) * 4;
    float4 t = vals[i];
    float4 r;
    r.x = (t.x - mu) * rstd * g[c0 + 0] + b[c0 + 0];
    r.y = (t.y - mu) * rstd * g[c0 + 1] + b[c0 + 1];
    r.z = (t.z - mu) * rstd * g[c0 + 2] + b[c0 + 2];
    r.w = (t.w - mu) * rstd * g[c0 + 3] + b[c0 + 3];
    *(float4*)(orow + c0) = r;
  }
}

// ===========================================================================
extern "C" void kernel_launch(void* const* d_in, const int* in_sizes, int n_in,
                              void* d_out, int out_size, void* d_ws,
                              size_t ws_size, hipStream_t stream) {
  const float* x      = (const float*)d_in[0];
  const int*   mask   = (const int*)d_in[1];
  const float* qkv_w  = (const float*)d_in[2];
  const float* qkv_b  = (const float*)d_in[3];
  const float* out_w  = (const float*)d_in[4];
  const float* out_b  = (const float*)d_in[5];
  const float* ffn_w1 = (const float*)d_in[6];
  const float* ffn_b1 = (const float*)d_in[7];
  const float* ffn_w2 = (const float*)d_in[8];
  const float* ffn_b2 = (const float*)d_in[9];
  const float* ln1_g  = (const float*)d_in[10];
  const float* ln1_b  = (const float*)d_in[11];
  const float* ln2_g  = (const float*)d_in[12];
  const float* ln2_b  = (const float*)d_in[13];

  if (ws_size < (size_t)64 * 1024 * 1024) return;  // need 64 MB scratch

  char* ws = (char*)d_ws;
  const size_t MB = 1024 * 1024;
  __bf16* qB = (__bf16*)(ws + 0 * MB);    // 8 MB
  __bf16* kB = (__bf16*)(ws + 8 * MB);    // 8 MB
  __bf16* vB = (__bf16*)(ws + 16 * MB);   // 8 MB
  __bf16* aB = (__bf16*)(ws + 24 * MB);   // 8 MB
  float*  xr = (float*)(ws + 32 * MB);    // 16 MB
  float*  x1 = (float*)(ws + 48 * MB);    // 16 MB
  __bf16* hB = (__bf16*)(ws + 0 * MB);    // 32 MB (reuses q/k/v/a)
  float*  y2 = (float*)(ws + 32 * MB);    // 16 MB (reuses xr)

  const int Mrows = Bz * Lq;  // 8192

  // 1) QKV projection -> q,k,v (bf16, head layout)
  {
    dim3 g((3 * Dm) / BN, Mrows / BM);
    gemm_kernel<0, float><<<g, 256, 0, stream>>>(
        x, qkv_w, qkv_b, nullptr, nullptr, qB, kB, vB, Mrows, 3 * Dm, Dm);
  }
  // 2) chunked local attention -> a (bf16, [B][L][D])
  {
    dim3 g(SCH / 128, Lq / SCH, Bz * Hh);
    attn_kernel<<<g, 256, 0, stream>>>(qB, kB, vB, mask, aB);
  }
  // 3) out projection + residual -> xr (f32)
  {
    dim3 g(Dm / BN, Mrows / BM);
    gemm_kernel<1, __bf16><<<g, 256, 0, stream>>>(
        aB, out_w, out_b, x, xr, nullptr, nullptr, nullptr, Mrows, Dm, Dm);
  }
  // 4) LayerNorm 1 -> x1
  ln_kernel<<<Mrows / 8, 256, 0, stream>>>(xr, ln1_g, ln1_b, x1);
  // 5) FFN1 + GELU -> h (bf16)
  {
    dim3 g(FFNd / BN, Mrows / BM);
    gemm_kernel<2, float><<<g, 256, 0, stream>>>(
        x1, ffn_w1, ffn_b1, nullptr, nullptr, hB, nullptr, nullptr,
        Mrows, FFNd, Dm);
  }
  // 6) FFN2 + residual -> y2 (f32)
  {
    dim3 g(Dm / BN, Mrows / BM);
    gemm_kernel<1, __bf16><<<g, 256, 0, stream>>>(
        hB, ffn_w2, ffn_b2, x1, y2, nullptr, nullptr, nullptr,
        Mrows, Dm, FFNd);
  }
  // 7) LayerNorm 2 -> out
  ln_kernel<<<Mrows / 8, 256, 0, stream>>>(y2, ln2_g, ln2_b, (float*)d_out);
}
